// TopologyRouter_44727789420745
// MI455X (gfx1250) — matrix operand order repaired and checked by reading, not compiled, not run on hardware
//
#include <hip/hip_runtime.h>
#include <math.h>

// ---------------------------------------------------------------------------
// Problem constants (B=4, S=2048, D=1024, N_GEO=3, K=8)
// ---------------------------------------------------------------------------
#define N_TOK   8192
#define DM      1024
#define KNN     8
#define EPSF    1e-8f

typedef _Float16 v16h __attribute__((ext_vector_type(16)));
typedef float    v8f  __attribute__((ext_vector_type(8)));
typedef _Float16 f16x4 __attribute__((ext_vector_type(4)));

union FragU { unsigned int u[8]; uint4 q[2]; v16h v; };

// CDNA5 async copy: global -> LDS, 16 bytes per lane, tracked by ASYNCcnt.
// lds_off is the precomputed 32-bit LDS byte offset (addr[31:0] of the
// generic pointer); gptr a 64-bit global address in a VGPR pair.
__device__ __forceinline__ void async_cp_b128(unsigned lds_off,
                                              const void* gptr) {
  asm volatile("global_load_async_to_lds_b128 %0, %1, off"
               :
               : "v"(lds_off), "v"(gptr)
               : "memory");
}
__device__ __forceinline__ void wait_async0() {
  asm volatile("s_wait_asynccnt 0x0" ::: "memory");
}

// Branch-free insert into descending sorted-9 list.
__device__ __forceinline__ void ins9(float (&t)[9], float v) {
#pragma unroll
  for (int j = 0; j < 9; ++j) {
    const float mx = fmaxf(t[j], v);
    v = fminf(t[j], v);
    t[j] = mx;
  }
}

// ---------------------------------------------------------------------------
// Kernel 1: per-row L2 normalize (fp32 -> f16) + content logits H @ Wc^T + bc
// One wave (32 lanes) per row. float4 coalesced loads, shfl_xor reductions.
// ---------------------------------------------------------------------------
__global__ __launch_bounds__(256) void norm_content_kernel(
    const float* __restrict__ H, const float* __restrict__ Wc,
    const float* __restrict__ bc, _Float16* __restrict__ Fh,
    float* __restrict__ cl) {
  const int wave = threadIdx.x >> 5;
  const int lane = threadIdx.x & 31;
  const int row  = blockIdx.x * 8 + wave;

  const float4* H4 = (const float4*)H + (size_t)row * (DM / 4);
  const float4* W4 = (const float4*)Wc;

  float4 h[8];
  float ss = 0.f, c0 = 0.f, c1 = 0.f, c2 = 0.f;
#pragma unroll
  for (int i = 0; i < 8; ++i) {
    const int idx = lane + 32 * i;          // 256 float4 per row
    float4 v = H4[idx];
    h[i] = v;
    ss += v.x * v.x + v.y * v.y + v.z * v.z + v.w * v.w;
    float4 w0 = W4[idx];
    float4 w1 = W4[256 + idx];
    float4 w2 = W4[512 + idx];
    c0 += v.x * w0.x + v.y * w0.y + v.z * w0.z + v.w * w0.w;
    c1 += v.x * w1.x + v.y * w1.y + v.z * w1.z + v.w * w1.w;
    c2 += v.x * w2.x + v.y * w2.y + v.z * w2.z + v.w * w2.w;
  }
#pragma unroll
  for (int off = 16; off >= 1; off >>= 1) {
    ss += __shfl_xor(ss, off, 32);
    c0 += __shfl_xor(c0, off, 32);
    c1 += __shfl_xor(c1, off, 32);
    c2 += __shfl_xor(c2, off, 32);
  }
  const float inv = 1.0f / (sqrtf(ss) + EPSF);

  f16x4* O = (f16x4*)Fh + (size_t)row * (DM / 4);
#pragma unroll
  for (int i = 0; i < 8; ++i) {
    const int idx = lane + 32 * i;
    f16x4 o = { (_Float16)(h[i].x * inv), (_Float16)(h[i].y * inv),
                (_Float16)(h[i].z * inv), (_Float16)(h[i].w * inv) };
    O[idx] = o;
  }
  if (lane == 0) {
    cl[(size_t)row * 3 + 0] = c0 + bc[0];
    cl[(size_t)row * 3 + 1] = c1 + bc[1];
    cl[(size_t)row * 3 + 2] = c2 + bc[2];
  }
}

// ---------------------------------------------------------------------------
// Kernel 2: fused  sim = F @ F^T  (f16 WMMA, f32 acc)  +  streaming top-9
// Block = 256 threads = 8 waves. Block owns 16 rows; per outer iteration it
// covers 512 columns (each wave: 4 C-tiles = 64 columns, 32 acc VGPRs).
// Per K-step (k=32): async-copy A(16x32) + B(512x32) chunks into LDS with
// global_load_async_to_lds_b128 / s_wait_asynccnt; source addresses are
// register-resident and bumped by +64 B per step; LDS dests are invariant.
// Each wave issues 4x v_wmma_f32_16x16x32_f16 off one shared A fragment.
// Finished 16x16 tiles feed a branch-free sorted-9 insert on all 32 lanes.
// ---------------------------------------------------------------------------
__global__ __launch_bounds__(256) void sim_topk_kernel(
    const _Float16* __restrict__ Fh, float* __restrict__ topo) {
  __shared__ _Float16 sAc[16 * 32];        //  1 KB : A chunk (16 rows x 32 k)
  __shared__ _Float16 sBc[512 * 32];       // 32 KB : B chunk (512 cols x 32 k)
  __shared__ float    simbuf[8][16][16];   //  8 KB : per-wave finished sims
  __shared__ float    topk[8][16][9];      // 4.5 KB: per-wave merged top-9

  const int tid  = threadIdx.x;
  const int wave = tid >> 5;
  const int lane = tid & 31;
  const int i0   = blockIdx.x * 16;        // this block's 16 rows
  const int Mrow = lane & 15;
  const int hi   = lane >> 4;

  float t9[9];
#pragma unroll
  for (int j = 0; j < 9; ++j) t9[j] = -2.0f;  // sims are >= -1

  const uint4* G = (const uint4*)Fh;       // 128 uint4 per row of F

  // --- loop-invariant staging descriptors (resident in VGPRs) ---
  unsigned ldsB[8];
#pragma unroll
  for (int v = 0; v < 8; ++v)
    ldsB[v] = (unsigned)(size_t)((uint4*)sBc + tid + 256 * v);
  const bool   aOn   = (tid & 3) == 0;     // 8 lanes per wave stage A
  const int    aIdx  = tid >> 2;           // 0..63: row=aIdx>>2, col=aIdx&3
  const uint4* gAb   = G + (size_t)(i0 + (aIdx >> 2)) * (DM / 8) + (aIdx & 3);
  const unsigned ldsA = (unsigned)(size_t)((uint4*)sAc + aIdx);

  for (int jb = 0; jb < N_TOK; jb += 512) {
    // per-strip B source base addresses; advanced by +64 B per K-step
    const uint4* gB[8];
#pragma unroll
    for (int v = 0; v < 8; ++v) {
      const int gi = tid + 256 * v;        // 0..2047: row=gi>>2, col=gi&3
      gB[v] = G + (size_t)(jb + (gi >> 2)) * (DM / 8) + (gi & 3);
    }
    const uint4* gA = gAb;

    v8f c[4] = {{}, {}, {}, {}};
    for (int ks = 0; ks < 32; ++ks) {      // K-steps of 32 halfs (64 B)
      __syncthreads();                     // previous chunk fully consumed
#pragma unroll
      for (int v = 0; v < 8; ++v) {
        async_cp_b128(ldsB[v], gB[v]);
        gB[v] += 4;                        // +64 B
      }
      if (aOn) async_cp_b128(ldsA, gA);
      gA += 4;
      wait_async0();
      __syncthreads();                     // chunk visible to all waves
      // --- A fragment (ISA 7.12.2 16x32 f16): shared by 4 tiles ---
      FragU a;
      {
        const _Float16* ap = sAc + Mrow * 32 + hi * 8;
        a.q[0] = *(const uint4*)ap;        // K pairs 0..7   (this half)
        a.q[1] = *(const uint4*)(ap + 16); // K pairs 16..23 (this half)
      }
#pragma unroll
      for (int t = 0; t < 4; ++t) {
        FragU b;
        // B 32x16 f16: lanes 0-15 K=0..7,16..23, lanes 16-31 K=8..15,24..31; N=lane%16
        const _Float16* bp =
            sBc + (wave * 64 + t * 16 + Mrow) * 32 + hi * 8;
        b.q[0] = *(const uint4*)bp;
        b.q[1] = *(const uint4*)(bp + 16);
        c[t] = __builtin_amdgcn_wmma_f32_16x16x32_f16(
            false, a.v, false, b.v, (short)0, c[t], false, false);
      }
    }
    // --- finished tiles: scatter via LDS, insert top-9 on all 32 lanes ---
#pragma unroll
    for (int t = 0; t < 4; ++t) {
#pragma unroll
      for (int r = 0; r < 8; ++r)
        simbuf[wave][r + (hi << 3)][Mrow] = c[t][r];  // M=r+8*hi, N=lane%16
      __builtin_amdgcn_wave_barrier();
#pragma unroll
      for (int n = 0; n < 8; ++n)          // lane handles row Mrow, half hi
        ins9(t9, simbuf[wave][Mrow][(hi << 3) + n]);
      __builtin_amdgcn_wave_barrier();
    }
  }

  // --- fold the two half-lists of each wave via shfl, then merge waves ---
  float p[9];
#pragma unroll
  for (int j = 0; j < 9; ++j) p[j] = __shfl(t9[j], lane + 16, 32);
  if (lane < 16) {
#pragma unroll
    for (int j = 0; j < 9; ++j) ins9(t9, p[j]);
#pragma unroll
    for (int j = 0; j < 9; ++j) topk[wave][lane][j] = t9[j];
  }
  __syncthreads();
  if (tid < 16) {
    float f9[9];
#pragma unroll
    for (int j = 0; j < 9; ++j) f9[j] = -2.0f;
    for (int w = 0; w < 8; ++w) {
#pragma unroll
      for (int j = 0; j < 9; ++j) ins9(f9, topk[w][tid][j]);
    }
    // f9[0] is the self-similarity (largest) -> dropped (top_k[:, 1:])
    float d[KNN];
    float mean = 0.f, mx = -1e30f;
#pragma unroll
    for (int j = 0; j < KNN; ++j) {
      d[j] = 1.0f - f9[j + 1];
      mean += d[j];
      mx = fmaxf(mx, d[j]);
    }
    mean *= (1.0f / KNN);
    float var = 0.f;
#pragma unroll
    for (int j = 0; j < KNN; ++j) {
      const float e = d[j] - mean;
      var += e * e;
    }
    const float stdv    = sqrtf(var * (1.0f / (KNN - 1)));  // ddof=1
    const float outlier = mx / (mean + EPSF);
    float* o = topo + (size_t)(i0 + tid) * 3;
    o[0] = stdv; o[1] = mean; o[2] = outlier;
  }
}

// ---------------------------------------------------------------------------
// Kernel 3: topo MLP (3->32->3), sigmoid mix, softmax; writes weights+logits
// ---------------------------------------------------------------------------
__global__ __launch_bounds__(256) void route_kernel(
    const float* __restrict__ cl, const float* __restrict__ topo,
    const float* __restrict__ W1, const float* __restrict__ b1,
    const float* __restrict__ W2, const float* __restrict__ b2,
    const float* __restrict__ alpha, float* __restrict__ out) {
  const int t = blockIdx.x * 256 + threadIdx.x;
  if (t >= N_TOK) return;
  const float f0 = topo[t * 3 + 0];
  const float f1 = topo[t * 3 + 1];
  const float f2 = topo[t * 3 + 2];
  float tl0 = b2[0], tl1 = b2[1], tl2 = b2[2];
#pragma unroll
  for (int i = 0; i < 32; ++i) {
    const float hh = fmaxf(
        W1[i * 3 + 0] * f0 + W1[i * 3 + 1] * f1 + W1[i * 3 + 2] * f2 + b1[i],
        0.f);
    tl0 += W2[i] * hh;
    tl1 += W2[32 + i] * hh;
    tl2 += W2[64 + i] * hh;
  }
  const float mix = 1.0f / (1.0f + expf(-alpha[0]));
  const float omix = 1.0f - mix;
  const float l0 = mix * cl[t * 3 + 0] + omix * tl0;
  const float l1 = mix * cl[t * 3 + 1] + omix * tl1;
  const float l2 = mix * cl[t * 3 + 2] + omix * tl2;
  const float m  = fmaxf(l0, fmaxf(l1, l2));
  const float e0 = expf(l0 - m), e1 = expf(l1 - m), e2 = expf(l2 - m);
  const float inv = 1.0f / (e0 + e1 + e2);
  out[t * 3 + 0] = e0 * inv;
  out[t * 3 + 1] = e1 * inv;
  out[t * 3 + 2] = e2 * inv;
  float* lo = out + (size_t)N_TOK * 3;
  lo[t * 3 + 0] = l0;
  lo[t * 3 + 1] = l1;
  lo[t * 3 + 2] = l2;
}

// ---------------------------------------------------------------------------
// Launch: inputs = H, Wc, bc, W1, b1, W2, b2, alpha (all fp32)
// Workspace: Fh (16 MB f16) | content_logits (96 KB) | topo_feats (96 KB)
// ---------------------------------------------------------------------------
extern "C" void kernel_launch(void* const* d_in, const int* in_sizes, int n_in,
                              void* d_out, int out_size, void* d_ws,
                              size_t ws_size, hipStream_t stream) {
  const float* H     = (const float*)d_in[0];
  const float* Wc    = (const float*)d_in[1];
  const float* bc    = (const float*)d_in[2];
  const float* W1    = (const float*)d_in[3];
  const float* b1    = (const float*)d_in[4];
  const float* W2    = (const float*)d_in[5];
  const float* b2    = (const float*)d_in[6];
  const float* alpha = (const float*)d_in[7];

  char* ws = (char*)d_ws;
  _Float16* Fh   = (_Float16*)ws;                                   // 16 MB
  float*    cl   = (float*)(ws + (size_t)N_TOK * DM * sizeof(_Float16));
  float*    topo = cl + (size_t)N_TOK * 3;
  float*    out  = (float*)d_out;

  norm_content_kernel<<<N_TOK / 8, 256, 0, stream>>>(H, Wc, bc, Fh, cl);
  sim_topk_kernel<<<N_TOK / 16, 256, 0, stream>>>(Fh, topo);
  route_kernel<<<(N_TOK + 255) / 256, 256, 0, stream>>>(cl, topo, W1, b1, W2,
                                                        b2, alpha, out);
}
